// TransChemGCNConv_15814069584045
// MI455X (gfx1250) — matrix operand order repaired and checked by reading, not compiled, not run on hardware
//
#include <hip/hip_runtime.h>

// ---------------------------------------------------------------------------
// CDNA5 (gfx1250) implementation of TransChemGCNConv-style graph conv.
//   Cin = Cout = 512 (hardcoded, matches reference), N derived from in_sizes.
//   GEMM: bf16x3 split-precision via v_wmma_f32_16x16x32_bf16 (wave32).
//   Edge phases: wave-per-edge coalesced float4 gathers + f32 global atomics.
// ---------------------------------------------------------------------------

typedef __attribute__((ext_vector_type(16))) __bf16 v16bf;
typedef __attribute__((ext_vector_type(2)))  __bf16 v2bf;
typedef __attribute__((ext_vector_type(8)))  float  v8f;

__device__ __forceinline__ v2bf load2bf(const __bf16* p) {
    unsigned int u = *(const unsigned int*)p;   // 4B-aligned (even k, 256B-aligned bases)
    v2bf r;
    __builtin_memcpy(&r, &u, sizeof(r));
    return r;
}

// ---------------- setup kernels ----------------

// deg_bond starts at 1.0 (self loop per node), deg_vin at 0; att2 = att_proj^2
__global__ void k_init(float* __restrict__ deg_b, float* __restrict__ deg_v,
                       float* __restrict__ att2, const float* __restrict__ att, int N) {
    int i = blockIdx.x * blockDim.x + threadIdx.x;
    if (i < N) { deg_b[i] = 1.0f; deg_v[i] = 0.0f; }
    if (i < 512) att2[i] = att[i] * att[i];
}

__global__ void k_count(const int* __restrict__ rows, int E, float* __restrict__ deg) {
    int e = blockIdx.x * blockDim.x + threadIdx.x;
    if (e < E) atomicAdd(deg + rows[e], 1.0f);
}

__global__ void k_finalize_deg(float* __restrict__ dinv, float* __restrict__ invdeg, int N) {
    int i = blockIdx.x * blockDim.x + threadIdx.x;
    if (i >= N) return;
    dinv[i]   = rsqrtf(dinv[i]);                  // deg >= 1 always (self loop)
    invdeg[i] = 1.0f / fmaxf(invdeg[i], 1.0f);    // clip(deg,1)
}

// f32 -> bf16 hi/lo split: f ~= hi + lo with |lo| <= ulp_bf16(f)/2
__global__ void k_convert(const float* __restrict__ src, __bf16* __restrict__ hi,
                          __bf16* __restrict__ lo, long long total) {
    long long i = (long long)blockIdx.x * blockDim.x + threadIdx.x;
    if (i >= total) return;
    float f = src[i];
    __bf16 h = (__bf16)f;
    hi[i] = h;
    lo[i] = (__bf16)(f - (float)h);
}

// concatenate [W_bond; W_vin; W_vconn] -> Wcat[1536][512] in bf16 hi/lo
__global__ void k_convert_w(const float* __restrict__ wb, const float* __restrict__ wv,
                            const float* __restrict__ wc, __bf16* __restrict__ hi,
                            __bf16* __restrict__ lo, int total /* 3*512*512 */) {
    int i = blockIdx.x * blockDim.x + threadIdx.x;
    if (i >= total) return;
    int which = i >> 18;            // / (512*512)
    int rem   = i & ((1 << 18) - 1);
    float f = (which == 0) ? wb[rem] : (which == 1) ? wv[rem] : wc[rem];
    __bf16 h = (__bf16)f;
    hi[i] = h;
    lo[i] = (__bf16)(f - (float)h);
}

// ---------------- WMMA GEMM: [N,512] x [512,1536]^T -> xb|xv|xc ----------------
// One wave computes a 16(M) x 64(N) strip. Block = 128 threads = 4 waves.
// Grid = (N/16, 6): 6*4 waves * 4 ntiles = 96 ntiles of 16 cols = 1536 cols.
__global__ void k_gemm_wmma(const __bf16* __restrict__ Xhi, const __bf16* __restrict__ Xlo,
                            const __bf16* __restrict__ Whi, const __bf16* __restrict__ Wlo,
                            float* __restrict__ xb, float* __restrict__ xv,
                            float* __restrict__ xc, int N) {
    const int lane   = threadIdx.x & 31;
    const int wave   = threadIdx.x >> 5;
    const int m0     = blockIdx.x * 16;
    const int ntile0 = (blockIdx.y * 4 + wave) * 4;   // 4 consecutive 16-col tiles

    v8f acc[4] = {};

    // A-matrix per-lane layout (16-bit A 16x32, ISA 7.12.2):
    //   lanes 0-15:  M=lane,    K = {0..7, 16..23}  (+kb)
    //   lanes 16-31: M=lane-16, K = {8..15, 24..31} (+kb)
    int mA = m0 + (lane & 15);
    if (mA >= N) mA = N - 1;                 // clamp (N is a multiple of 16 in practice)
    const int kA_lane = (lane < 16) ? 0 : 8;
    // B-matrix per-lane layout (B[k][n] = Wcat[n][k]):
    //   lanes 0-15:  n=lane,    K={0..7, 16..23} (+kb);  lanes 16-31: n=lane-16, K={8..15, 24..31} (+kb)
    const int nB      = lane & 15;
    const int kB_lane = (lane < 16) ? 0 : 8;

    const __bf16* __restrict__ xhiRow = Xhi + (long long)mA * 512;
    const __bf16* __restrict__ xloRow = Xlo + (long long)mA * 512;

    for (int kb = 0; kb < 512; kb += 32) {
        v16bf a_hi, a_lo;
#pragma unroll
        for (int j = 0; j < 8; ++j) {
            int k = kb + ((j < 4) ? (kA_lane + 2 * j) : (16 + kA_lane + 2 * (j - 4)));
            v2bf h = load2bf(xhiRow + k);
            v2bf l = load2bf(xloRow + k);
            a_hi[2 * j] = h[0]; a_hi[2 * j + 1] = h[1];
            a_lo[2 * j] = l[0]; a_lo[2 * j + 1] = l[1];
        }
#pragma unroll
        for (int t = 0; t < 4; ++t) {
            const int cog = (ntile0 + t) * 16 + nB;   // 0..1535
            const __bf16* whiRow = Whi + (long long)cog * 512;
            const __bf16* wloRow = Wlo + (long long)cog * 512;
            v16bf b_hi, b_lo;
#pragma unroll
            for (int j = 0; j < 8; ++j) {
                int k = kb + ((j < 4) ? (kB_lane + 2 * j) : (16 + kB_lane + 2 * (j - 4)));
                v2bf h = load2bf(whiRow + k);
                v2bf l = load2bf(wloRow + k);
                b_hi[2 * j] = h[0]; b_hi[2 * j + 1] = h[1];
                b_lo[2 * j] = l[0]; b_lo[2 * j + 1] = l[1];
            }
            // bf16x3 split: hi*hi + hi*lo + lo*hi  (lo*lo term ~2^-32, dropped)
            acc[t] = __builtin_amdgcn_wmma_f32_16x16x32_bf16(false, a_hi, false, b_hi,
                                                             (short)0, acc[t], false, false);
            acc[t] = __builtin_amdgcn_wmma_f32_16x16x32_bf16(false, a_hi, false, b_lo,
                                                             (short)0, acc[t], false, false);
            acc[t] = __builtin_amdgcn_wmma_f32_16x16x32_bf16(false, a_lo, false, b_hi,
                                                             (short)0, acc[t], false, false);
        }
    }

    // C/D layout: lanes 0-15 -> N=lane, M=r; lanes 16-31 -> N=lane-16, M=r+8
    const int mbase = m0 + ((lane >> 4) << 3);
#pragma unroll
    for (int t = 0; t < 4; ++t) {
        const int cog = (ntile0 + t) * 16 + nB;
        const int sel = cog >> 9;          // which output matrix
        const int co  = cog & 511;
        float* dst = (sel == 0) ? xb : (sel == 1) ? xv : xc;
#pragma unroll
        for (int r = 0; r < 8; ++r) {
            int m = mbase + r;
            if (m < N) dst[(long long)m * 512 + co] = acc[t][r];
        }
    }
}

// ---------------- output init: bias + self-loop term ----------------
__global__ void k_init_out(const float* __restrict__ xb, const float* __restrict__ dinv,
                           const float* __restrict__ bias, float* __restrict__ out,
                           long long total) {
    long long i = (long long)blockIdx.x * blockDim.x + threadIdx.x;
    if (i >= total) return;
    int n = (int)(i >> 9);
    int c = (int)(i & 511);
    float d = dinv[n];
    out[i] = bias[c] + d * d * xb[i];
}

// ---------------- edge scatter kernels: one wave per edge ----------------
__global__ void k_bond(const int* __restrict__ eb, int E, const float* __restrict__ xb,
                       const float* __restrict__ dinv, float* __restrict__ out) {
    const int gw   = blockIdx.x * 8 + (threadIdx.x >> 5);
    const int lane = threadIdx.x & 31;
    if (gw >= E) return;
    const int r = eb[gw], c = eb[E + gw];
    const float s = dinv[r] * dinv[c];
    const float4* __restrict__ src = (const float4*)(xb + (long long)c * 512);
    float* __restrict__ dst = out + (long long)r * 512;
#pragma unroll
    for (int i = 0; i < 4; ++i) {
        const int q = i * 32 + lane;
        float4 v = src[q];
        atomicAdd(dst + 4 * q + 0, v.x * s);
        atomicAdd(dst + 4 * q + 1, v.y * s);
        atomicAdd(dst + 4 * q + 2, v.z * s);
        atomicAdd(dst + 4 * q + 3, v.w * s);
    }
}

__global__ void k_intra(const int* __restrict__ ev, int E, const float* __restrict__ xv,
                        const float* __restrict__ invdeg, float* __restrict__ out) {
    const int gw   = blockIdx.x * 8 + (threadIdx.x >> 5);
    const int lane = threadIdx.x & 31;
    if (gw >= E) return;
    const int r = ev[gw], c = ev[E + gw];
    const float s = invdeg[r];
    const float4* __restrict__ src = (const float4*)(xv + (long long)c * 512);
    float* __restrict__ dst = out + (long long)r * 512;
#pragma unroll
    for (int i = 0; i < 4; ++i) {
        const int q = i * 32 + lane;
        float4 v = src[q];
        atomicAdd(dst + 4 * q + 0, v.x * s);
        atomicAdd(dst + 4 * q + 1, v.y * s);
        atomicAdd(dst + 4 * q + 2, v.z * s);
        atomicAdd(dst + 4 * q + 3, v.w * s);
    }
}

__global__ void k_conn(const int* __restrict__ ec, int E, const float* __restrict__ xc,
                       const float* __restrict__ att2, float* __restrict__ out) {
    const int gw   = blockIdx.x * 8 + (threadIdx.x >> 5);
    const int lane = threadIdx.x & 31;
    if (gw >= E) return;
    const int r = ec[gw], c = ec[E + gw];
    const float4* __restrict__ pr = (const float4*)(xc + (long long)r * 512);
    const float4* __restrict__ pc = (const float4*)(xc + (long long)c * 512);
    const float4* __restrict__ pa = (const float4*)att2;

    float acc = 0.0f;
    float4 vc[4];
#pragma unroll
    for (int i = 0; i < 4; ++i) {
        const int q = i * 32 + lane;
        float4 a  = pa[q];
        float4 xi = pr[q];
        float4 xj = pc[q];
        vc[i] = xj;
        acc += a.x * xi.x * xj.x + a.y * xi.y * xj.y + a.z * xi.z * xj.z + a.w * xi.w * xj.w;
    }
    // wave32 all-reduce
#pragma unroll
    for (int off = 16; off > 0; off >>= 1) acc += __shfl_xor(acc, off, 32);
    const float alpha = 1.0f / (1.0f + __expf(-acc));

    float* __restrict__ dst = out + (long long)r * 512;
#pragma unroll
    for (int i = 0; i < 4; ++i) {
        const int q = i * 32 + lane;
        atomicAdd(dst + 4 * q + 0, vc[i].x * alpha);
        atomicAdd(dst + 4 * q + 1, vc[i].y * alpha);
        atomicAdd(dst + 4 * q + 2, vc[i].z * alpha);
        atomicAdd(dst + 4 * q + 3, vc[i].w * alpha);
    }
}

// ---------------------------------------------------------------------------

extern "C" void kernel_launch(void* const* d_in, const int* in_sizes, int n_in,
                              void* d_out, int out_size, void* d_ws, size_t ws_size,
                              hipStream_t stream) {
    const float* x    = (const float*)d_in[0];
    const int*   eb   = (const int*)d_in[1];
    const int*   ev   = (const int*)d_in[2];
    const int*   ec   = (const int*)d_in[3];
    const float* Wb   = (const float*)d_in[4];
    const float* Wv   = (const float*)d_in[5];
    const float* Wc   = (const float*)d_in[6];
    const float* att  = (const float*)d_in[7];
    const float* bias = (const float*)d_in[8];
    float*       out  = (float*)d_out;

    const long long N  = in_sizes[0] / 512;
    const int Eb = in_sizes[1] / 2;
    const int Ev = in_sizes[2] / 2;
    const int Ec = in_sizes[3] / 2;
    const long long NC = N * 512;

    // workspace carve-out (256B aligned)
    char* w = (char*)d_ws;
    auto carve = [&](size_t bytes) {
        char* p = w;
        w += (bytes + 255) & ~(size_t)255;
        return (void*)p;
    };
    float*  xb     = (float*)carve((size_t)NC * 4);
    float*  xv     = (float*)carve((size_t)NC * 4);
    float*  xc     = (float*)carve((size_t)NC * 4);
    __bf16* Xhi    = (__bf16*)carve((size_t)NC * 2);
    __bf16* Xlo    = (__bf16*)carve((size_t)NC * 2);
    __bf16* Whi    = (__bf16*)carve((size_t)3 * 512 * 512 * 2);
    __bf16* Wlo    = (__bf16*)carve((size_t)3 * 512 * 512 * 2);
    float*  dinv   = (float*)carve((size_t)N * 4);   // holds deg_bond, finalized in place
    float*  invdeg = (float*)carve((size_t)N * 4);   // holds deg_vin, finalized in place
    float*  att2   = (float*)carve(512 * 4);

    const int B = 256;
    const long long initN = (N > 512 ? N : 512);

    k_init<<<(unsigned)((initN + B - 1) / B), B, 0, stream>>>(dinv, invdeg, att2, att, (int)N);
    k_count<<<(Eb + B - 1) / B, B, 0, stream>>>(eb, Eb, dinv);
    k_count<<<(Ev + B - 1) / B, B, 0, stream>>>(ev, Ev, invdeg);
    k_finalize_deg<<<(unsigned)((N + B - 1) / B), B, 0, stream>>>(dinv, invdeg, (int)N);

    k_convert<<<(unsigned)((NC + B - 1) / B), B, 0, stream>>>(x, Xhi, Xlo, NC);
    k_convert_w<<<(3 * 512 * 512 + B - 1) / B, B, 0, stream>>>(Wb, Wv, Wc, Whi, Wlo,
                                                               3 * 512 * 512);

    dim3 gg((unsigned)((N + 15) / 16), 6);
    k_gemm_wmma<<<gg, 128, 0, stream>>>(Xhi, Xlo, Whi, Wlo, xb, xv, xc, (int)N);

    k_init_out<<<(unsigned)((NC + B - 1) / B), B, 0, stream>>>(xb, dinv, bias, out, NC);

    k_bond<<<(Eb + 7) / 8, B, 0, stream>>>(eb, Eb, xb, dinv, out);
    k_intra<<<(Ev + 7) / 8, B, 0, stream>>>(ev, Ev, xv, invdeg, out);
    k_conn<<<(Ec + 7) / 8, B, 0, stream>>>(ec, Ec, xc, att2, out);
}
